// MetaJanusRRPRAM_50440095924329
// MI455X (gfx1250) — matrix operand order repaired and checked by reading, not compiled, not run on hardware
//
#include <hip/hip_runtime.h>
#include <hip/hip_bf16.h>

#define E_ 768
#define H_ 12
#define D_ 64
#define T_ 1024
#define L_ 4
#define M_ 3072
#define V_ 256
#define B_ 2

typedef __attribute__((ext_vector_type(16))) __bf16 v16bf;
typedef __attribute__((ext_vector_type(8)))  __bf16 v8bf;
typedef __attribute__((ext_vector_type(8)))  float  v8f;

// ---------------------------------------------------------------------------
// Elementwise / reduction kernels
// ---------------------------------------------------------------------------

__global__ void embed_kernel(const int* __restrict__ idx, const float* __restrict__ tok,
                             const float* __restrict__ pos, float* __restrict__ x) {
    int row = blockIdx.x;              // b*T + t
    int t = row % T_;
    int tokid = idx[row];
    const float* te = tok + (size_t)tokid * E_;
    const float* pe = pos + (size_t)t * E_;
    float* xr = x + (size_t)row * E_;
    for (int e = threadIdx.x; e < E_; e += blockDim.x)
        xr[e] = te[e] + pe[e];
}

__global__ void rmsnorm_kernel(const float* __restrict__ x, const float* __restrict__ w,
                               float* __restrict__ y) {
    __shared__ float red[256];
    int row = blockIdx.x;
    const float* xr = x + (size_t)row * E_;
    float s = 0.f;
    for (int e = threadIdx.x; e < E_; e += 256) { float v = xr[e]; s += v * v; }
    red[threadIdx.x] = s; __syncthreads();
    for (int o = 128; o > 0; o >>= 1) {
        if (threadIdx.x < o) red[threadIdx.x] += red[threadIdx.x + o];
        __syncthreads();
    }
    float scale = rsqrtf(red[0] / (float)E_ + 1e-5f);
    float* yr = y + (size_t)row * E_;
    for (int e = threadIdx.x; e < E_; e += 256) yr[e] = xr[e] * scale * w[e];
}

// r[b,h,t] = sum_e xn[b,t,e] * wr[h,e,t]   (wr already offset to layer l)
__global__ void rscore_kernel(const float* __restrict__ xn, const float* __restrict__ wr,
                              float* __restrict__ r) {
    int gid = blockIdx.x * 256 + threadIdx.x;   // (b*H + h)*T + t
    int t  = gid % T_;
    int bh = gid / T_;
    int h  = bh % H_;
    int b  = bh / H_;
    const float* xr  = xn + (size_t)(b * T_ + t) * E_;
    const float* wrh = wr + (size_t)h * E_ * T_ + t;
    float s = 0.f;
    for (int e = 0; e < E_; ++e) s += xr[e] * wrh[(size_t)e * T_];
    r[gid] = s;
}

// s[b,t] = dot(xn[b,t], eb[b,t]) / sqrt(E)
__global__ void sdot_kernel(const float* __restrict__ xn, const float* __restrict__ eb,
                            float* __restrict__ sout) {
    __shared__ float red[256];
    int row = blockIdx.x;
    const float* a = xn + (size_t)row * E_;
    const float* b = eb + (size_t)row * E_;
    float s = 0.f;
    for (int e = threadIdx.x; e < E_; e += 256) s += a[e] * b[e];
    red[threadIdx.x] = s; __syncthreads();
    for (int o = 128; o > 0; o >>= 1) {
        if (threadIdx.x < o) red[threadIdx.x] += red[threadIdx.x + o];
        __syncthreads();
    }
    if (threadIdx.x == 0) sout[row] = red[0] * rsqrtf((float)E_);
}

// Online-softmax prefix scan: RRP scores are query-independent so
// out[q] = (sum_{k<=q} e^{s_k} rv[k]) / (sum_{k<=q} e^{s_k}).  One block per (b,h).
__global__ void rrp_scan_kernel(const float* __restrict__ r, const float* __restrict__ rv,
                                float* __restrict__ out) {
    int bh = blockIdx.x; int b = bh / H_; int h = bh % H_;
    int d = threadIdx.x;                      // 0..63
    const float* rr = r + (size_t)bh * T_;
    const float scale = 0.125f;               // 1/sqrt(D)
    float m = -INFINITY, den = 0.f, num = 0.f;
    for (int k = 0; k < T_; ++k) {
        float sk = rr[k] * scale;
        float mn = fmaxf(m, sk);
        float c = __expf(m - mn);
        float p = __expf(sk - mn);
        size_t off = (size_t)(b * T_ + k) * (H_ * D_) + h * D_ + d;
        num = num * c + p * rv[off];
        den = den * c + p;
        m = mn;
        out[off] = num / den;
    }
}

// pj[b,q,k] = softmax_k( s_q * s_k  for k<=q )   (zeros above diagonal)
__global__ void pj_kernel(const float* __restrict__ s, float* __restrict__ pj) {
    __shared__ float red[256];
    int row = blockIdx.x; int b = row / T_; int q = row % T_;
    const float* sb = s + (size_t)b * T_;
    float sq = sb[q];
    float mx = -INFINITY;
    for (int k = threadIdx.x; k <= q; k += 256) mx = fmaxf(mx, sq * sb[k]);
    red[threadIdx.x] = mx; __syncthreads();
    for (int o = 128; o > 0; o >>= 1) {
        if (threadIdx.x < o) red[threadIdx.x] = fmaxf(red[threadIdx.x], red[threadIdx.x + o]);
        __syncthreads();
    }
    mx = red[0]; __syncthreads();
    float sum = 0.f;
    for (int k = threadIdx.x; k <= q; k += 256) sum += __expf(sq * sb[k] - mx);
    red[threadIdx.x] = sum; __syncthreads();
    for (int o = 128; o > 0; o >>= 1) {
        if (threadIdx.x < o) red[threadIdx.x] += red[threadIdx.x + o];
        __syncthreads();
    }
    float inv = 1.f / red[0];
    float* pr = pj + (size_t)row * T_;
    for (int k = threadIdx.x; k < T_; k += 256)
        pr[k] = (k <= q) ? __expf(sq * sb[k] - mx) * inv : 0.f;
}

// out = g0[h]*rrp + g1[h]*jan, g = softmax(gate[h,:]) over 2
__global__ void combine_kernel(const float* __restrict__ rrp, const float* __restrict__ jan,
                               const float* __restrict__ gate, float* __restrict__ out) {
    int row = blockIdx.x;
    size_t base = (size_t)row * (H_ * D_);
    for (int i = threadIdx.x; i < H_ * D_; i += 256) {
        int h = i >> 6;
        float a = gate[h * 2 + 0], c = gate[h * 2 + 1];
        float mg = fmaxf(a, c);
        float ea = __expf(a - mg), ec = __expf(c - mg);
        float inv = 1.f / (ea + ec);
        out[base + i] = (ea * inv) * rrp[base + i] + (ec * inv) * jan[base + i];
    }
}

__global__ void silu_mul_kernel(float* __restrict__ g, const float* __restrict__ u, size_t n) {
    size_t i = (size_t)blockIdx.x * 256 + threadIdx.x;
    if (i < n) { float x = g[i]; g[i] = (x / (1.f + __expf(-x))) * u[i]; }
}

// ---------------------------------------------------------------------------
// bf16 WMMA GEMM: C[M,N] = A[M,K] @ op(W)  (+C if ACCUM)
//   TRANSW=true : W is [N,K] row-major (C = A @ W^T), fully coalesced along K
//   TRANSW=false: W is [K,N] row-major (C = A @ W),   LDS transpose on stage
// 256 threads (8 waves), 128x128 block tile, BK=64; each wave owns a 32x64
// subtile: 2 A-fragments x 4 B-fragments -> 8 v_wmma_f32_16x16x32_bf16 per
// 32-K step from only 6 fragment loads (high wmma/ds density).
// ---------------------------------------------------------------------------
template<bool TRANSW, bool ACCUM>
__global__ void gemm_wmma_kernel(const float* __restrict__ A, const float* __restrict__ W,
                                 float* __restrict__ C, int Mdim, int Ndim, int Kdim) {
    __shared__ __bf16 As[128][72];   // [m][k], stride 72 keeps 16B-aligned fragments
    __shared__ __bf16 Ws[128][72];   // [n][k]
    const int tid  = threadIdx.x;
    const int m0   = blockIdx.y * 128;
    const int n0   = blockIdx.x * 128;
    const int lane = tid & 31;
    const int wave = tid >> 5;
    const int wm   = (wave & 3) * 32;    // waves 4x2 over (m,n)
    const int wn   = (wave >> 2) * 64;
    const int hm   = lane & 15;          // row (A) / col (B) within 16x16 tile
    const int hk   = lane >> 4;          // lane half selects K sub-range
    v8f acc[2][4] = {};

    for (int k0 = 0; k0 < Kdim; k0 += 64) {
        // Prefetch next K-tile of A while we stage this one (global_prefetch_b8).
        if (k0 + 64 < Kdim) {
            int pr = tid >> 1;                       // 128 rows, 2 lines each
            int pc = k0 + 64 + (tid & 1) * 32;
            __builtin_prefetch(A + (size_t)(m0 + pr) * Kdim + pc, 0, 3);
            if (TRANSW)
                __builtin_prefetch(W + (size_t)(n0 + pr) * Kdim + pc, 0, 3);
        }
        // ---- stage A tile (128 rows x 64 k), fp32 -> bf16
        #pragma unroll
        for (int i = 0; i < 8; ++i) {
            int lin = i * 256 + tid;
            int r = lin >> 4, cg = (lin & 15) * 4;
            float4 v = *(const float4*)(A + (size_t)(m0 + r) * Kdim + k0 + cg);
            As[r][cg + 0] = (__bf16)v.x; As[r][cg + 1] = (__bf16)v.y;
            As[r][cg + 2] = (__bf16)v.z; As[r][cg + 3] = (__bf16)v.w;
        }
        // ---- stage W tile as [n][k] (128 n x 64 k)
        if (TRANSW) {
            #pragma unroll
            for (int i = 0; i < 8; ++i) {
                int lin = i * 256 + tid;
                int r = lin >> 4, cg = (lin & 15) * 4;
                float4 v = *(const float4*)(W + (size_t)(n0 + r) * Kdim + k0 + cg);
                Ws[r][cg + 0] = (__bf16)v.x; Ws[r][cg + 1] = (__bf16)v.y;
                Ws[r][cg + 2] = (__bf16)v.z; Ws[r][cg + 3] = (__bf16)v.w;
            }
        } else {
            #pragma unroll
            for (int i = 0; i < 8; ++i) {
                int lin = i * 256 + tid;
                int kk = lin >> 5, cg = (lin & 31) * 4;  // coalesced along n, scatter to LDS
                float4 v = *(const float4*)(W + (size_t)(k0 + kk) * Ndim + n0 + cg);
                Ws[cg + 0][kk] = (__bf16)v.x; Ws[cg + 1][kk] = (__bf16)v.y;
                Ws[cg + 2][kk] = (__bf16)v.z; Ws[cg + 3][kk] = (__bf16)v.w;
            }
        }
        __syncthreads();

        #pragma unroll
        for (int ks = 0; ks < 2; ++ks) {
            const int kb = ks * 32;
            v16bf af[2], bf[4];
            #pragma unroll
            for (int i = 0; i < 2; ++i) {
                // A fragment: lane half interleaves K (0..7|8..15 then 16..23|24..31)
                v8bf* ap = (v8bf*)&af[i];
                ap[0] = *(const v8bf*)&As[wm + i * 16 + hm][kb + hk * 8];
                ap[1] = *(const v8bf*)&As[wm + i * 16 + hm][kb + 16 + hk * 8];
            }
            #pragma unroll
            for (int j = 0; j < 4; ++j) {
                // B fragment: lane half interleaves K (0..7|8..15 then 16..23|24..31)
                v8bf* bp = (v8bf*)&bf[j];
                bp[0] = *(const v8bf*)&Ws[wn + j * 16 + hm][kb + hk * 8];
                bp[1] = *(const v8bf*)&Ws[wn + j * 16 + hm][kb + 16 + hk * 8];
            }
            #pragma unroll
            for (int mi = 0; mi < 2; ++mi)
                #pragma unroll
                for (int ni = 0; ni < 4; ++ni)
                    acc[mi][ni] = __builtin_amdgcn_wmma_f32_16x16x32_bf16(
                        false, af[mi], false, bf[ni], (short)0, acc[mi][ni], false, false);
        }
        __syncthreads();
    }

    #pragma unroll
    for (int mi = 0; mi < 2; ++mi)
        #pragma unroll
        for (int ni = 0; ni < 4; ++ni)
            #pragma unroll
            for (int r = 0; r < 8; ++r) {
                int row = m0 + wm + mi * 16 + hk * 8 + r;
                int col = n0 + wn + ni * 16 + hm;
                size_t o = (size_t)row * Ndim + col;
                float v = acc[mi][ni][r];
                if (ACCUM) C[o] += v; else C[o] = v;
            }
}

// ---------------------------------------------------------------------------

extern "C" void kernel_launch(void* const* d_in, const int* in_sizes, int n_in,
                              void* d_out, int out_size, void* d_ws, size_t ws_size,
                              hipStream_t stream) {
    const int*   idx   = (const int*)  d_in[0];
    const float* tok   = (const float*)d_in[1];
    const float* pos   = (const float*)d_in[2];
    const float* rms1  = (const float*)d_in[3];
    const float* wr    = (const float*)d_in[4];
    const float* wvr   = (const float*)d_in[5];
    const float* wj    = (const float*)d_in[6];
    const float* gate  = (const float*)d_in[7];
    const float* wo    = (const float*)d_in[8];
    const float* rms2  = (const float*)d_in[9];
    const float* wg    = (const float*)d_in[10];
    const float* wu    = (const float*)d_in[11];
    const float* wd    = (const float*)d_in[12];
    const float* rmsf  = (const float*)d_in[13];
    const float* headw = (const float*)d_in[14];
    float* out = (float*)d_out;

    float* ws = (float*)d_ws;
    const size_t BT  = (size_t)B_ * T_;        // 2048
    const size_t BTE = BT * E_;
    float* X  = ws;  ws += BTE;
    float* XN = ws;  ws += BTE;
    float* T0 = ws;  ws += BTE;                // rv, then jan_out
    float* T1 = ws;  ws += BTE;                // echo, then combined
    float* T2 = ws;  ws += BTE;                // echo_back, then rrp_out
    float* Rr = ws;  ws += (size_t)B_ * H_ * T_;
    float* Ss = ws;  ws += BT;
    float* PJ = ws;  ws += BT * T_;
    float* GB = ws;  ws += BT * M_;
    float* UB = ws;  ws += BT * M_;

    dim3 blk256(256), blk64(64);
    const int rows = (int)BT;

    embed_kernel<<<rows, blk256, 0, stream>>>(idx, tok, pos, X);

    for (int l = 0; l < L_; ++l) {
        const float* wr_l   = wr  + (size_t)l * H_ * E_ * T_;
        const float* wvr_l  = wvr + (size_t)l * (H_ * D_) * E_;
        const float* wj_l   = wj  + (size_t)l * E_ * E_;
        const float* wo_l   = wo  + (size_t)l * E_ * (H_ * D_);
        const float* wg_l   = wg  + (size_t)l * M_ * E_;
        const float* wu_l   = wu  + (size_t)l * M_ * E_;
        const float* wd_l   = wd  + (size_t)l * E_ * M_;
        const float* gate_l = gate+ (size_t)l * H_ * 2;

        rmsnorm_kernel<<<rows, blk256, 0, stream>>>(X, rms1 + (size_t)l * E_, XN);
        // rv = xn @ wvr^T
        gemm_wmma_kernel<true, false><<<dim3((H_ * D_) / 128, rows / 128), blk256, 0, stream>>>(
            XN, wvr_l, T0, rows, H_ * D_, E_);
        // r[b,h,t]
        rscore_kernel<<<(B_ * H_ * T_) / 256, blk256, 0, stream>>>(XN, wr_l, Rr);
        // echo = xn @ wj^T
        gemm_wmma_kernel<true, false><<<dim3(E_ / 128, rows / 128), blk256, 0, stream>>>(
            XN, wj_l, T1, rows, E_, E_);
        // echo_back = echo @ wj
        gemm_wmma_kernel<false, false><<<dim3(E_ / 128, rows / 128), blk256, 0, stream>>>(
            T1, wj_l, T2, rows, E_, E_);
        // s[b,t]
        sdot_kernel<<<rows, blk256, 0, stream>>>(XN, T2, Ss);
        // rrp_out -> T2 (prefix online-softmax scan; overwrites echo_back)
        rrp_scan_kernel<<<B_ * H_, blk64, 0, stream>>>(Rr, T0, T2);
        // pj softmax matrix
        pj_kernel<<<rows, blk256, 0, stream>>>(Ss, PJ);
        // jan_out = pj @ echo  (per batch) -> T0 (overwrites rv)
        for (int b = 0; b < B_; ++b)
            gemm_wmma_kernel<false, false><<<dim3((H_ * D_) / 128, T_ / 128), blk256, 0, stream>>>(
                PJ + (size_t)b * T_ * T_, T1 + (size_t)b * T_ * (H_ * D_),
                T0 + (size_t)b * T_ * (H_ * D_), T_, H_ * D_, T_);
        // gated combine -> T1 (overwrites echo)
        combine_kernel<<<rows, blk256, 0, stream>>>(T2, T0, gate_l, T1);
        // x += combined @ wo^T
        gemm_wmma_kernel<true, true><<<dim3(E_ / 128, rows / 128), blk256, 0, stream>>>(
            T1, wo_l, X, rows, E_, H_ * D_);
        // MLP
        rmsnorm_kernel<<<rows, blk256, 0, stream>>>(X, rms2 + (size_t)l * E_, XN);
        gemm_wmma_kernel<true, false><<<dim3(M_ / 128, rows / 128), blk256, 0, stream>>>(
            XN, wg_l, GB, rows, M_, E_);
        gemm_wmma_kernel<true, false><<<dim3(M_ / 128, rows / 128), blk256, 0, stream>>>(
            XN, wu_l, UB, rows, M_, E_);
        silu_mul_kernel<<<(int)((BT * M_) / 256), blk256, 0, stream>>>(GB, UB, BT * M_);
        gemm_wmma_kernel<true, true><<<dim3(E_ / 128, rows / 128), blk256, 0, stream>>>(
            GB, wd_l, X, rows, E_, M_);
    }

    rmsnorm_kernel<<<rows, blk256, 0, stream>>>(X, rmsf, XN);
    gemm_wmma_kernel<true, false><<<dim3(V_ / 128, rows / 128), blk256, 0, stream>>>(
        XN, headw, out, rows, V_, E_);
}
